// SetAbstraction_5128190951729
// MI455X (gfx1250) — matrix operand order repaired and checked by reading, not compiled, not run on hardware
//
#include <hip/hip_runtime.h>
#include <hip/hip_bf16.h>

// ---------------------------------------------------------------------------
// PointNet++ SetAbstraction for MI455X (gfx1250, wave32).
// conv2 / conv3 / fc1 run on v_wmma_f32_16x16x32_bf16 (f32 accumulate).
// One wave per 16-point n-tile holds the normalized B panel in registers and
// reuses it across 8 M-tiles (8 WMMAs per B load).  Weight (A) loads are
// float4 -> global_load_b128.  BN folded to per-channel scale/shift;
// normalize+ReLU+bf16-cvt fused into the consumer GEMM's B load.
// No atomics anywhere -> deterministic across graph replays.
// ---------------------------------------------------------------------------

#define BATCH   32
#define NPTS    2048
#define NCENT   128
#define NGRP    32
#define PTOT    (BATCH * NCENT * NGRP)   // 131072 grouped points
#define R2      0.04f
#define BN_EPS  1e-5f
#define FC1_K   32768
#define FC1_S   16                        // split-K slices for fc1

typedef __bf16 v16bf __attribute__((ext_vector_type(16)));
typedef float  v8f   __attribute__((ext_vector_type(8)));

// ---------------------------------------------------------------------------
// 1) Farthest point sampling: one block per batch, points + dist in LDS.
// ---------------------------------------------------------------------------
__global__ void fps_kernel(const float* __restrict__ x, float* __restrict__ cent) {
  __shared__ float px[NPTS], py[NPTS], pz[NPTS], dist[NPTS];
  __shared__ float rv[256];
  __shared__ int   ri[256];
  const int b = blockIdx.x, t = threadIdx.x;
  const float* xb = x + (size_t)b * 3 * NPTS;
  for (int n = t; n < NPTS; n += 256) {
    px[n] = xb[n]; py[n] = xb[NPTS + n]; pz[n] = xb[2 * NPTS + n];
  }
  __syncthreads();
  const float fx = px[0], fy = py[0], fz = pz[0];
  for (int n = t; n < NPTS; n += 256) {
    float dx = px[n] - fx, dy = py[n] - fy, dz = pz[n] - fz;
    dist[n] = dx * dx + dy * dy + dz * dz;
  }
  if (t == 0) {
    cent[((size_t)b * 3 + 0) * NCENT] = fx;
    cent[((size_t)b * 3 + 1) * NCENT] = fy;
    cent[((size_t)b * 3 + 2) * NCENT] = fz;
  }
  __syncthreads();
  for (int c = 1; c < NCENT; ++c) {
    float bv = -1.0f; int bi = 0;
    for (int n = t; n < NPTS; n += 256) {
      float v = dist[n];
      if (v > bv) { bv = v; bi = n; }          // strict > keeps first max
    }
    rv[t] = bv; ri[t] = bi;
    __syncthreads();
    for (int s = 128; s > 0; s >>= 1) {
      if (t < s) {
        if (rv[t + s] > rv[t] || (rv[t + s] == rv[t] && ri[t + s] < ri[t])) {
          rv[t] = rv[t + s]; ri[t] = ri[t + s];
        }
      }
      __syncthreads();
    }
    const int idx = ri[0];
    const float cx = px[idx], cy = py[idx], cz = pz[idx];
    if (t == 0) {
      cent[((size_t)b * 3 + 0) * NCENT + c] = cx;
      cent[((size_t)b * 3 + 1) * NCENT + c] = cy;
      cent[((size_t)b * 3 + 2) * NCENT + c] = cz;
    }
    for (int n = t; n < NPTS; n += 256) {
      float dx = px[n] - cx, dy = py[n] - cy, dz = pz[n] - cz;
      float nd = dx * dx + dy * dy + dz * dz;
      if (nd < dist[n]) dist[n] = nd;
    }
    __syncthreads();
  }
}

// ---------------------------------------------------------------------------
// 2) Radius grouping: one wave per (b,c). Ballot+popcount prefix scan picks
//    the first K in-radius indices in ascending order (== sort-then-take-K).
//    g layout: [3][PTOT], p = (b*NCENT+c)*NGRP + k  (matches einsum bdck).
// ---------------------------------------------------------------------------
__global__ void group_kernel(const float* __restrict__ x,
                             const float* __restrict__ cent,
                             float* __restrict__ g) {
  __shared__ int sidx[8][NGRP];
  const int wave = blockIdx.x * 8 + (threadIdx.x >> 5);
  const int lane = threadIdx.x & 31;
  int* widx = sidx[threadIdx.x >> 5];
  const int b = wave / NCENT, c = wave % NCENT;
  const float* xb = x + (size_t)b * 3 * NPTS;
  const float cx = cent[((size_t)b * 3 + 0) * NCENT + c];
  const float cy = cent[((size_t)b * 3 + 1) * NCENT + c];
  const float cz = cent[((size_t)b * 3 + 2) * NCENT + c];
  int cnt = 0;
  for (int n0 = 0; n0 < NPTS && cnt < NGRP; n0 += 32) {
    const int n = n0 + lane;
    float dx = xb[n] - cx, dy = xb[NPTS + n] - cy, dz = xb[2 * NPTS + n] - cz;
    float sq = dx * dx + dy * dy + dz * dz;
    bool within = (sq <= R2);
    unsigned m = __builtin_amdgcn_ballot_w32(within);
    int pos = cnt + __builtin_popcount(m & ((1u << lane) - 1u));
    if (within && pos < NGRP) widx[pos] = n;
    cnt += __builtin_popcount(m);                 // wave-uniform
  }
  __syncthreads();
  const int first = (cnt > 0) ? widx[0] : 0;
  const int idx = (lane < cnt) ? widx[lane] : first;
  const size_t p = (size_t)wave * NGRP + lane;    // lane == k, NGRP == 32
  g[p]                    = xb[idx] - cx;
  g[(size_t)PTOT + p]     = xb[NPTS + idx] - cy;
  g[2 * (size_t)PTOT + p] = xb[2 * NPTS + idx] - cz;
}

// ---------------------------------------------------------------------------
// 3) conv1 (3->64): K=3 too small for WMMA; plain FMAs, coalesced over p.
// ---------------------------------------------------------------------------
__global__ void conv1_kernel(const float* __restrict__ g,
                             const float* __restrict__ w1,
                             const float* __restrict__ b1,
                             float* __restrict__ y1) {
  const int p = blockIdx.x * 256 + threadIdx.x;
  const float a0 = g[p], a1 = g[PTOT + p], a2 = g[2 * PTOT + p];
#pragma unroll 8
  for (int o = 0; o < 64; ++o) {
    y1[(size_t)o * PTOT + p] =
        w1[o * 3] * a0 + w1[o * 3 + 1] * a1 + w1[o * 3 + 2] * a2 + b1[o];
  }
}

// ---------------------------------------------------------------------------
// BN statistics: one block per channel, deterministic tree reduction.
// Folds (gamma,beta,mean,var) into scale/shift:  y_norm = y*scale + shift.
// ---------------------------------------------------------------------------
__global__ void bn_stats_kernel(const float* __restrict__ y,
                                const float* __restrict__ gam,
                                const float* __restrict__ bet,
                                float* __restrict__ scale,
                                float* __restrict__ shift, int P) {
  __shared__ float s1[256], s2[256];
  const int ch = blockIdx.x, t = threadIdx.x;
  const float* row = y + (size_t)ch * P;
  float a = 0.f, b = 0.f;
  for (int p = t; p < P; p += 256) { float v = row[p]; a += v; b += v * v; }
  s1[t] = a; s2[t] = b;
  __syncthreads();
  for (int s = 128; s > 0; s >>= 1) {
    if (t < s) { s1[t] += s1[t + s]; s2[t] += s2[t + s]; }
    __syncthreads();
  }
  if (t == 0) {
    float m = s1[0] / (float)P;
    float v = s2[0] / (float)P - m * m;
    float sc = gam[ch] * rsqrtf(v + BN_EPS);
    scale[ch] = sc;
    shift[ch] = bet[ch] - m * sc;
  }
}

// ---------------------------------------------------------------------------
// WMMA layouts (CDNA5 ISA 7.12.2), wave32:
//  A (16x32 bf16): lane half h, elem i -> M = lane&15,
//                  K = i + 16*h (order used here, not the ISA table)
//                  => one contiguous 16-float run: [16h .. 16h+15]
//  B (32x16 bf16): elem i -> K = i + 16*h (same order as A), N = lane&15
//  D (16x16 f32):  reg r  -> M = r + 8*h,  N = lane&15
// ---------------------------------------------------------------------------
__device__ __forceinline__ v16bf load_a_bf16(const float* __restrict__ wrow) {
  // wrow already offset by +16*half; four float4 -> global_load_b128.
  float4 w0 = *(const float4*)(wrow);
  float4 w1 = *(const float4*)(wrow + 4);
  float4 w2 = *(const float4*)(wrow + 8);
  float4 w3 = *(const float4*)(wrow + 12);
  v16bf a;
  a[0]  = (__bf16)w0.x; a[1]  = (__bf16)w0.y; a[2]  = (__bf16)w0.z; a[3]  = (__bf16)w0.w;
  a[4]  = (__bf16)w1.x; a[5]  = (__bf16)w1.y; a[6]  = (__bf16)w1.z; a[7]  = (__bf16)w1.w;
  a[8]  = (__bf16)w2.x; a[9]  = (__bf16)w2.y; a[10] = (__bf16)w2.z; a[11] = (__bf16)w2.w;
  a[12] = (__bf16)w3.x; a[13] = (__bf16)w3.y; a[14] = (__bf16)w3.z; a[15] = (__bf16)w3.w;
  return a;
}

// conv2/conv3 GEMM: one wave per 16-point n-tile computes a 128-row panel
// (8 M-tiles) so the normalized B operand is loaded/converted once and fed to
// 8 WMMAs.  conv3 (M=256) uses blockIdx.y to split into two 128-row panels.
__global__ void conv_gemm_kernel(const float* __restrict__ W,
                                 const float* __restrict__ X,
                                 const float* __restrict__ sc,
                                 const float* __restrict__ sh,
                                 const float* __restrict__ bias,
                                 float* __restrict__ Y, int Ktot) {
  const int wave = blockIdx.x * (blockDim.x >> 5) + (threadIdx.x >> 5);
  const int n0   = wave << 4;                    // exact grid: no guard needed
  const int moff = blockIdx.y << 7;              // 128-row panel
  const int lane = threadIdx.x & 31;
  const int half = lane >> 4;
  const int l16  = lane & 15;
  v8f acc[8];
#pragma unroll
  for (int m = 0; m < 8; ++m) acc[m] = (v8f){0.f,0.f,0.f,0.f,0.f,0.f,0.f,0.f};
  for (int k0 = 0; k0 < Ktot; k0 += 32) {
    v16bf b;
#pragma unroll
    for (int i = 0; i < 16; ++i) {
      int k = k0 + i + (half << 4);
      float xv = X[(size_t)k * PTOT + n0 + l16];
      xv = xv * sc[k] + sh[k];
      xv = xv > 0.f ? xv : 0.f;                  // fused BN + ReLU
      b[i] = (__bf16)xv;
    }
#pragma unroll
    for (int m = 0; m < 8; ++m) {
      const float* wrow =
          W + (size_t)(moff + m * 16 + l16) * Ktot + k0 + (half << 4);
      v16bf a = load_a_bf16(wrow);
      acc[m] = __builtin_amdgcn_wmma_f32_16x16x32_bf16(false, a, false, b,
                                                       (short)0, acc[m],
                                                       false, false);
    }
  }
#pragma unroll
  for (int m = 0; m < 8; ++m) {
#pragma unroll
    for (int r = 0; r < 8; ++r) {
      int mm = moff + m * 16 + r + (half << 3);
      Y[(size_t)mm * PTOT + n0 + l16] = acc[m][r] + bias[mm];
    }
  }
}

// ---------------------------------------------------------------------------
// 4) BN+ReLU+max over NGRP, transpose to feature-major bf16 for fc1 B-operand.
//    feature f = ch*NCENT + cc  (matches h.reshape(-1, 256*C)).
// ---------------------------------------------------------------------------
__global__ void maxpool_kernel(const float* __restrict__ y3,
                               const float* __restrict__ sc3,
                               const float* __restrict__ sh3,
                               __bf16* __restrict__ pooledT) {
  const int gid = blockIdx.x * 256 + threadIdx.x;      // 32768 * 32
  const int n = gid & 31, f = gid >> 5;
  const int ch = f / NCENT, cc = f % NCENT;
  const float* base = y3 + (size_t)ch * PTOT + ((size_t)n * NCENT + cc) * NGRP;
  const float s = sc3[ch], h = sh3[ch];
  float mx = 0.f;                                      // relu output >= 0
#pragma unroll 8
  for (int k = 0; k < NGRP; ++k) {
    float v = base[k] * s + h;
    v = v > 0.f ? v : 0.f;
    mx = v > mx ? v : mx;
  }
  pooledT[(size_t)f * 32 + n] = (__bf16)mx;
}

// ---------------------------------------------------------------------------
// 5) fc1 WMMA with split-K=16 (deterministic partials, no atomics).
//    134 MB wf1 stream dominates: each f32 weight read by both n-tile waves via b128,
//    converted to bf16 in-register.
// ---------------------------------------------------------------------------
__global__ void fc1_gemm_kernel(const float* __restrict__ W,
                                const __bf16* __restrict__ Xb,
                                float* __restrict__ partials) {
  const int wave  = blockIdx.x * (blockDim.x >> 5) + (threadIdx.x >> 5);
  const int mt    = 1024 >> 4, nt = 2;                 // 64 x 2 tiles
  const int slice = wave / (mt * nt);
  const int t     = wave % (mt * nt);
  const int m0 = (t % mt) << 4, n0 = (t / mt) << 4;
  const int lane = threadIdx.x & 31;
  const int half = lane >> 4, l16 = lane & 15;
  const int kPer = FC1_K / FC1_S;
  const int kbeg = slice * kPer;
  v8f acc = {0.f, 0.f, 0.f, 0.f, 0.f, 0.f, 0.f, 0.f};
  const float* wrow = W + (size_t)(m0 + l16) * FC1_K + (half << 4);
  for (int k0 = kbeg; k0 < kbeg + kPer; k0 += 32) {
    v16bf a = load_a_bf16(wrow + k0);
    v16bf b;
#pragma unroll
    for (int i = 0; i < 16; ++i) {
      int k = k0 + i + (half << 4);
      b[i] = Xb[(size_t)k * 32 + n0 + l16];
    }
    acc = __builtin_amdgcn_wmma_f32_16x16x32_bf16(false, a, false, b,
                                                  (short)0, acc, false, false);
  }
#pragma unroll
  for (int r = 0; r < 8; ++r) {
    int m = m0 + r + (half << 3);
    partials[((size_t)slice * 1024 + m) * 32 + n0 + l16] = acc[r];
  }
}

__global__ void fc1_reduce_kernel(const float* __restrict__ partials,
                                  const float* __restrict__ bf1,
                                  float* __restrict__ yf1) {
  const int gid = blockIdx.x * 256 + threadIdx.x;      // 1024 * 32
  const int n = gid & 31, f = gid >> 5;
  float acc = bf1[f];
#pragma unroll
  for (int s = 0; s < FC1_S; ++s)
    acc += partials[((size_t)s * 1024 + f) * 32 + n];
  yf1[(size_t)f * 32 + n] = acc;
}

// ---------------------------------------------------------------------------
// 6) fc2 (1024->256) scalar, input normalized on the fly; feature-major out.
// ---------------------------------------------------------------------------
__global__ void fc2_kernel(const float* __restrict__ W,
                           const float* __restrict__ bias,
                           const float* __restrict__ Xf,
                           const float* __restrict__ sc,
                           const float* __restrict__ sh,
                           float* __restrict__ Y) {
  const int gid = blockIdx.x * 256 + threadIdx.x;      // 256 * 32
  const int n = gid & 31, o = gid >> 5;
  const float* wr = W + (size_t)o * 1024;
  float acc = bias[o];
  for (int k = 0; k < 1024; ++k) {
    float v = Xf[(size_t)k * 32 + n] * sc[k] + sh[k];
    v = v > 0.f ? v : 0.f;
    acc += wr[k] * v;
  }
  Y[(size_t)o * 32 + n] = acc;
}

// 7) fc3 (256->256) scalar, writes final output batch-major (B, 256).
__global__ void fc3_kernel(const float* __restrict__ wf3,
                           const float* __restrict__ bf3,
                           const float* __restrict__ yf2,
                           const float* __restrict__ sc,
                           const float* __restrict__ sh,
                           float* __restrict__ out) {
  const int gid = blockIdx.x * 256 + threadIdx.x;      // 32 * 256
  const int n = gid >> 8, o = gid & 255;
  float acc = bf3[o];
  for (int k = 0; k < 256; ++k) {
    float v = yf2[(size_t)k * 32 + n] * sc[k] + sh[k];
    v = v > 0.f ? v : 0.f;
    acc += wf3[(size_t)o * 256 + k] * v;
  }
  out[(size_t)n * 256 + o] = acc;
}

// ---------------------------------------------------------------------------
extern "C" void kernel_launch(void* const* d_in, const int* in_sizes, int n_in,
                              void* d_out, int out_size, void* d_ws, size_t ws_size,
                              hipStream_t stream) {
  const float* x   = (const float*)d_in[0];
  const float* w1  = (const float*)d_in[1];
  const float* b1  = (const float*)d_in[2];
  const float* g1  = (const float*)d_in[3];
  const float* be1 = (const float*)d_in[4];
  const float* w2  = (const float*)d_in[5];
  const float* b2  = (const float*)d_in[6];
  const float* g2  = (const float*)d_in[7];
  const float* be2 = (const float*)d_in[8];
  const float* w3  = (const float*)d_in[9];
  const float* b3  = (const float*)d_in[10];
  const float* g3  = (const float*)d_in[11];
  const float* be3 = (const float*)d_in[12];
  const float* wf1 = (const float*)d_in[13];
  const float* bf1 = (const float*)d_in[14];
  const float* g4  = (const float*)d_in[15];
  const float* be4 = (const float*)d_in[16];
  const float* wf2 = (const float*)d_in[17];
  const float* bf2 = (const float*)d_in[18];
  const float* g5  = (const float*)d_in[19];
  const float* be5 = (const float*)d_in[20];
  const float* wf3 = (const float*)d_in[21];
  const float* bf3 = (const float*)d_in[22];

  char* ws = (char*)d_ws;
  size_t off = 0;
  auto alloc = [&](size_t bytes) -> void* {
    void* p = ws + off;
    off = (off + bytes + 255) & ~(size_t)255;
    return p;
  };
  float*  cent     = (float*)alloc((size_t)BATCH * 3 * NCENT * 4);
  float*  grouped  = (float*)alloc((size_t)3 * PTOT * 4);
  float*  y1       = (float*)alloc((size_t)64  * PTOT * 4);
  float*  y2       = (float*)alloc((size_t)128 * PTOT * 4);
  float*  y3       = (float*)alloc((size_t)256 * PTOT * 4);
  __bf16* pooledT  = (__bf16*)alloc((size_t)FC1_K * 32 * 2);
  float*  partials = (float*)alloc((size_t)FC1_S * 1024 * 32 * 4);
  float*  yf1      = (float*)alloc((size_t)1024 * 32 * 4);
  float*  yf2      = (float*)alloc((size_t)256 * 32 * 4);
  float*  sc1 = (float*)alloc(64 * 4),   *sh1 = (float*)alloc(64 * 4);
  float*  sc2 = (float*)alloc(128 * 4),  *sh2 = (float*)alloc(128 * 4);
  float*  sc3 = (float*)alloc(256 * 4),  *sh3 = (float*)alloc(256 * 4);
  float*  sc4 = (float*)alloc(1024 * 4), *sh4 = (float*)alloc(1024 * 4);
  float*  sc5 = (float*)alloc(256 * 4),  *sh5 = (float*)alloc(256 * 4);

  fps_kernel<<<BATCH, 256, 0, stream>>>(x, cent);
  group_kernel<<<(BATCH * NCENT) / 8, 256, 0, stream>>>(x, cent, grouped);
  conv1_kernel<<<PTOT / 256, 256, 0, stream>>>(grouped, w1, b1, y1);
  bn_stats_kernel<<<64, 256, 0, stream>>>(y1, g1, be1, sc1, sh1, PTOT);
  // conv2: M=128 -> one 128-row panel (grid.y = 1); 8192 n-tiles / 8 waves
  conv_gemm_kernel<<<dim3((PTOT >> 4) / 8, 1), 256, 0, stream>>>(
      w2, y1, sc1, sh1, b2, y2, 64);
  bn_stats_kernel<<<128, 256, 0, stream>>>(y2, g2, be2, sc2, sh2, PTOT);
  // conv3: M=256 -> two 128-row panels (grid.y = 2)
  conv_gemm_kernel<<<dim3((PTOT >> 4) / 8, 2), 256, 0, stream>>>(
      w3, y2, sc2, sh2, b3, y3, 128);
  bn_stats_kernel<<<256, 256, 0, stream>>>(y3, g3, be3, sc3, sh3, PTOT);
  maxpool_kernel<<<(FC1_K * 32) / 256, 256, 0, stream>>>(y3, sc3, sh3, pooledT);
  fc1_gemm_kernel<<<(64 * 2 * FC1_S) / 8, 256, 0, stream>>>(wf1, pooledT, partials);
  fc1_reduce_kernel<<<(1024 * 32) / 256, 256, 0, stream>>>(partials, bf1, yf1);
  bn_stats_kernel<<<1024, 256, 0, stream>>>(yf1, g4, be4, sc4, sh4, 32);
  fc2_kernel<<<(256 * 32) / 256, 256, 0, stream>>>(wf2, bf2, yf1, sc4, sh4, yf2);
  bn_stats_kernel<<<256, 256, 0, stream>>>(yf2, g5, be5, sc5, sh5, 32);
  fc3_kernel<<<(32 * 256) / 256, 256, 0, stream>>>(wf3, bf3, yf2, sc5, sh5,
                                                   (float*)d_out);
}
